// DelayedXOR_DH_SNN_2637109920458
// MI455X (gfx1250) — matrix operand order repaired and checked by reading, not compiled, not run on hardware
//
#include <hip/hip_runtime.h>

// ---------------------------------------------------------------------------
// DH-SNN scan + readout for gfx1250 (wave32, WMMA bf16 16x16x32, f32 accum)
// ---------------------------------------------------------------------------

typedef __attribute__((ext_vector_type(16))) __bf16        v16bf;
typedef __attribute__((ext_vector_type(8)))  float         v8f;
typedef __attribute__((ext_vector_type(4)))  unsigned int  v4u;

struct V4Pair { v4u lo; v4u hi; };

__device__ __forceinline__ v16bf frag_from(const v4u* p0, const v4u* p1) {
  V4Pair t; t.lo = *p0; t.hi = *p1;          // 2x b128 loads
  return __builtin_bit_cast(v16bf, t);
}

// one-instruction pack: {bf16(hi), bf16(lo)} via v_perm_b32 (truncation)
__device__ __forceinline__ unsigned int pack_bf2(float lo, float hi) {
  return __builtin_amdgcn_perm(__float_as_uint(hi), __float_as_uint(lo),
                               0x07060302u);
}

// RNE conversion for the one-time acc writeback
__device__ __forceinline__ unsigned int f2bf_rne(float f) {
  unsigned int u = __float_as_uint(f);
  return (u + 0x7FFFu + ((u >> 16) & 1u)) >> 16;
}

__device__ __forceinline__ float sigmoidf_(float x) {
  return 1.0f / (1.0f + __expf(-x));
}

// Problem dims
#define BB 256
#define TT 512
#define II 512
#define HH 1024
#define NBR 4
#define HBR 256
#define OO 128

// ---------------------------------------------------------------------------
// Kernel 1: full time scan. Each block: 32 (batch) x 64 (hidden) tile.
// 256 threads = 8 waves; wave (wm, wn) owns the 16x16 C tile at
// rows b0+wm*16.., cols c0+wn*16.. . State d,v,s,acc live in VGPRs for all T.
// W (B-matrix) fragments are loop-invariant -> held in 128 VGPRs per wave,
// so the hot loop only touches LDS for A fragments (2x b128 per WMMA).
// x_t is double-buffered in LDS; t+1 is prefetched into registers while the
// WMMAs for t run, so the per-step global latency is hidden.
// ---------------------------------------------------------------------------
__global__ __launch_bounds__(256)
void snn_scan_kernel(const float* __restrict__ x,       // (B,T,I)
                     const float* __restrict__ v0,      // (B,H)
                     const float* __restrict__ Wb,      // (NB,I,HB)
                     const float* __restrict__ tau_n,   // (NB,HB)
                     const float* __restrict__ tau_m,   // (H,)
                     unsigned short* __restrict__ acc_bf16) // (B,H) bf16 out
{
  __shared__ __align__(16) unsigned int xlds[2][32 * 256];     // 2 x 32KB
  __shared__ __align__(16) unsigned int wlds[4 * 16 * 32 * 8]; // 64KB swizzled W

  const int tid  = threadIdx.x;
  const int lane = tid & 31;
  const int half = lane >> 4;
  const int w    = tid >> 5;       // wave 0..7
  const int wm   = w & 1;          // M tile index (2)
  const int wn   = w >> 1;         // N tile index (4)
  const int c0   = blockIdx.x * 64;  // hidden column base (within one branch)
  const int b0   = blockIdx.y * 32;  // batch row base
  const int nb0  = c0 / HBR;
  const int hb0  = c0 % HBR;

  // ---- one-time: stage W tile (I x 64) into fragment-ready bf16 layout ----
  // layout (uints): [ntile(4)][kb(16)][lane(32)][8]; per lane 16 bf16 = B-frag
  // element e of lane L: n = ntile*16 + L%16, K = kb*32 + (L/16)*16 + e
  for (int idx = tid; idx < 4 * 16 * 32 * 8; idx += 256) {
    int j  = idx & 7;
    int ln = (idx >> 3) & 31;
    int kb = (idx >> 8) & 15;
    int nt = idx >> 12;
    int n  = nt * 16 + (ln & 15);
    int K  = kb * 32 + ((ln >> 4) << 4) + 2 * j;
    const float* wp = Wb + (size_t)nb0 * II * HBR + (size_t)K * HBR + hb0 + n;
    wlds[idx] = pack_bf2(wp[0], wp[HBR]);   // K and K+1, same column
  }
  __syncthreads();

  // ---- hoist this wave's 16 B-fragments into registers (loop-invariant) ----
  v16bf bfr[16];
#pragma unroll
  for (int kb = 0; kb < 16; ++kb) {
    const unsigned int* brow = &wlds[(((wn * 16 + kb) * 32) + lane) * 8];
    bfr[kb] = frag_from((const v4u*)brow, (const v4u*)(brow + 4));
  }

  // per-lane (per-column) decay constants
  const int colL   = lane & 15;
  const int hcol   = c0 + wn * 16 + colL;
  const float alpha = sigmoidf_(tau_m[hcol]);
  const float beta  = sigmoidf_(tau_n[nb0 * HBR + hb0 + wn * 16 + colL]);
  const float oma = 1.0f - alpha, omb = 1.0f - beta;

  // recurrent state in VGPRs (C-layout: VGPR r, lane L -> row r + 8*(L/16))
  v8f d, v, s, acc;
#pragma unroll
  for (int r = 0; r < 8; ++r) {
    int row = b0 + wm * 16 + r + half * 8;
    d[r] = 0.0f; s[r] = 0.0f; acc[r] = 0.0f;
    v[r] = v0[(size_t)row * HH + hcol];
  }

  const int srow = tid >> 3;         // staging: 8 threads per row (0..31)
  const int si0  = (tid & 7) * 64;   // 64 consecutive floats each
  const float* xbase = x + (size_t)(b0 + srow) * TT * II + si0;

  // prologue: prefetch + stage t = 0 into buffer 0
  float4 g[16];
#pragma unroll
  for (int q = 0; q < 16; ++q) g[q] = *(const float4*)(xbase + q * 4);
#pragma unroll
  for (int q = 0; q < 16; ++q) {
    uint2 p; p.x = pack_bf2(g[q].x, g[q].y); p.y = pack_bf2(g[q].z, g[q].w);
    *(uint2*)&xlds[0][srow * 256 + ((si0 + q * 4) >> 1)] = p;
  }

  for (int t = 0; t < TT; ++t) {
    __syncthreads();   // buffer (t&1) fully staged, buffer ((t+1)&1) free

    // prefetch t+1 into registers; these loads fly under the WMMAs below
    const int tn = (t + 1 < TT) ? (t + 1) : t;
    const float* xr = xbase + (size_t)tn * II;
#pragma unroll
    for (int q = 0; q < 16; ++q) g[q] = *(const float4*)(xr + q * 4);

    // branch_in tile: C = x_t_tile @ W_tile  (16x16, K=512)
    const unsigned int* cur = xlds[t & 1];
    v8f c = {};
#pragma unroll
    for (int kb = 0; kb < 16; ++kb) {
      const unsigned int* arow =
          &cur[(wm * 16 + colL) * 256 + kb * 16 + half * 8];
      v16bf af = frag_from((const v4u*)arow, (const v4u*)(arow + 4));
      c = __builtin_amdgcn_wmma_f32_16x16x32_bf16(
              false, af, false, bfr[kb], (short)0, c, false, false);
    }

    // LIF update (element-wise, per accumulator register)
    const float m = (t >= TT / 2) ? 1.0f : 0.0f;
#pragma unroll
    for (int r = 0; r < 8; ++r) {
      d[r]   = beta * d[r] + omb * c[r];            // dendrite filter
      v[r]   = alpha * v[r] + oma * d[r] - s[r];    // V_TH = 1 (reset term)
      s[r]   = (v[r] > 1.0f) ? 1.0f : 0.0f;         // spike
      acc[r] += s[r] * m;                           // integrate 2nd half
    }

    // stage t+1 into the other buffer (harmless rewrite on the last step)
    unsigned int* nxt = xlds[(t + 1) & 1];
#pragma unroll
    for (int q = 0; q < 16; ++q) {
      uint2 p; p.x = pack_bf2(g[q].x, g[q].y); p.y = pack_bf2(g[q].z, g[q].w);
      *(uint2*)&nxt[srow * 256 + ((si0 + q * 4) >> 1)] = p;
    }
  }

  // write spike counts as bf16 for the readout GEMM
#pragma unroll
  for (int r = 0; r < 8; ++r) {
    int row = b0 + wm * 16 + r + half * 8;
    acc_bf16[(size_t)row * HH + hcol] = (unsigned short)f2bf_rne(acc[r]);
  }
}

// ---------------------------------------------------------------------------
// Kernel 2: out = acc @ W_out + b.  (256x1024)x(1024x128), bf16 WMMA.
// Blocks: 64x64 tile, 16 waves, each one 16x16 C tile, K-loop of 32 x 32.
// A-fragments load straight from global (acc is already bf16, row-major).
// ---------------------------------------------------------------------------
__global__ __launch_bounds__(512)
void out_gemm_kernel(const unsigned short* __restrict__ acc_bf16, // (B,H) bf16
                     const float* __restrict__ Wout,              // (H,O)
                     const float* __restrict__ bout,              // (O,)
                     float* __restrict__ out)                     // (B,O)
{
  __shared__ __align__(16) unsigned int wlds[4 * 32 * 32 * 8]; // 128KB swizzled

  const int tid  = threadIdx.x;
  const int lane = tid & 31;
  const int half = lane >> 4;
  const int w    = tid >> 5;
  const int wm   = w & 3;
  const int wn   = w >> 2;
  const int o0   = blockIdx.x * 64;
  const int b0   = blockIdx.y * 64;

  // stage W_out tile (1024 x 64) fragment-ready: [ntile(4)][kb(32)][lane][8]
  for (int idx = tid; idx < 4 * 32 * 32 * 8; idx += 512) {
    int j  = idx & 7;
    int ln = (idx >> 3) & 31;
    int kb = (idx >> 8) & 31;
    int nt = idx >> 13;
    int n  = o0 + nt * 16 + (ln & 15);
    int K  = kb * 32 + ((ln >> 4) << 4) + 2 * j;
    wlds[idx] = pack_bf2(Wout[(size_t)K * OO + n], Wout[(size_t)(K + 1) * OO + n]);
  }
  __syncthreads();

  v8f c = {};
  const unsigned short* arow =
      acc_bf16 + (size_t)(b0 + wm * 16 + (lane & 15)) * HH;
#pragma unroll 4
  for (int kb = 0; kb < 32; ++kb) {
    const v4u* a0 = (const v4u*)(arow + kb * 32 + half * 16);
    const v4u* a1 = (const v4u*)(arow + kb * 32 + half * 16 + 8);
    v16bf af = frag_from(a0, a1);
    const unsigned int* brow = &wlds[(((wn * 32 + kb) * 32) + lane) * 8];
    v16bf bf = frag_from((const v4u*)brow, (const v4u*)(brow + 4));
    c = __builtin_amdgcn_wmma_f32_16x16x32_bf16(
            false, af, false, bf, (short)0, c, false, false);
  }

  const int col  = o0 + wn * 16 + (lane & 15);
  const float bias = bout[col];
#pragma unroll
  for (int r = 0; r < 8; ++r) {
    int row = b0 + wm * 16 + r + half * 8;
    out[(size_t)row * OO + col] = c[r] + bias;
  }
}

// ---------------------------------------------------------------------------
extern "C" void kernel_launch(void* const* d_in, const int* in_sizes, int n_in,
                              void* d_out, int out_size, void* d_ws, size_t ws_size,
                              hipStream_t stream) {
  const float* x    = (const float*)d_in[0];  // (256,512,512)
  const float* v0   = (const float*)d_in[1];  // (256,1024)
  const float* Wb   = (const float*)d_in[2];  // (4,512,256)
  const float* taun = (const float*)d_in[3];  // (4,256)
  const float* taum = (const float*)d_in[4];  // (1024,)
  const float* Wout = (const float*)d_in[5];  // (1024,128)
  const float* bout = (const float*)d_in[6];  // (128,)
  unsigned short* accb = (unsigned short*)d_ws;  // (256,1024) bf16 scratch

  dim3 g1(HH / 64, BB / 32);   // 16 x 8 = 128 blocks
  snn_scan_kernel<<<g1, dim3(256), 0, stream>>>(x, v0, Wb, taun, taum, accb);

  dim3 g2(OO / 64, BB / 64);   // 2 x 4
  out_gemm_kernel<<<g2, dim3(512), 0, stream>>>(accb, Wout, bout, (float*)d_out);
}
